// DeepBKT_51908974739859
// MI455X (gfx1250) — matrix operand order repaired and checked by reading, not compiled, not run on hardware
//
#include <hip/hip_runtime.h>
#include <cstddef>

// Problem constants
#define BB   32
#define SS   512
#define DDm  512
#define HHn  8
#define DKK  64
#define LLn  4
#define FFF  2048
#define MROWS (BB*SS)          // 16384
#define NEGV (-1e32f)

typedef __bf16 bf16_t;
typedef __attribute__((ext_vector_type(16))) __bf16 v16bf;
typedef __attribute__((ext_vector_type(8)))  __bf16 v8bf;
typedef __attribute__((ext_vector_type(8)))  float  v8f;
typedef __attribute__((ext_vector_type(4)))  int    v4i_t;

#if __has_builtin(__builtin_amdgcn_global_load_async_to_lds_b128) && \
    __has_builtin(__builtin_amdgcn_s_wait_asynccnt)
#define ASYNC_LDS 1
#endif

static __device__ __forceinline__ bf16_t to_bf16(float f) {
  unsigned int u = __builtin_bit_cast(unsigned int, f);
  u = (u + 0x7FFFu + ((u >> 16) & 1u)) >> 16;
  unsigned short s = (unsigned short)u;
  return __builtin_bit_cast(bf16_t, s);
}

// Copy 16 bytes global -> LDS (async on gfx1250 if builtins available).
static __device__ __forceinline__ void copy16_g2l(bf16_t* ldst, const bf16_t* gsrc) {
#ifdef ASYNC_LDS
  // two-step casts: generic -> AS-qualified void* (addrspacecast), then pointee cast
  __attribute__((address_space(1))) const void* gv =
      (__attribute__((address_space(1))) const void*)gsrc;
  __attribute__((address_space(3))) void* lv =
      (__attribute__((address_space(3))) void*)ldst;
  __builtin_amdgcn_global_load_async_to_lds_b128(
      (__attribute__((address_space(1))) v4i_t*)gv,
      (__attribute__((address_space(3))) v4i_t*)lv, 0, 0);
#else
  *(v8bf*)ldst = *(const v8bf*)gsrc;
#endif
}
static __device__ __forceinline__ void async_fence() {
#ifdef ASYNC_LDS
  __builtin_amdgcn_s_wait_asynccnt(0);
#endif
}

// A fragment (16x32 bf16, M rows x K cols), tile row-major [m][k], ld in elems.
// ISA: lanes 0-15 => M=lane, K={0..7,16..23}; lanes 16-31 => K={8..15,24..31}
static __device__ __forceinline__ v16bf load_frag_a(const bf16_t* tile, int ld,
                                                    int mbase, int kbase) {
  int lane = threadIdx.x & 31;
  const bf16_t* p = tile + (size_t)(mbase + (lane & 15)) * ld + kbase + ((lane >> 4) << 3);
  v8bf lo = *(const v8bf*)p;
  v8bf hi = *(const v8bf*)(p + 16);
  return __builtin_shufflevector(lo, hi, 0,1,2,3,4,5,6,7,8,9,10,11,12,13,14,15);
}

// B fragment (32x16 bf16, K rows x N cols), tile stored N-major [n][k], ld in elems.
// ISA: lanes 0-15 => N=lane, K={0..7,16..23}; lanes 16-31 => K={8..15,24..31}
static __device__ __forceinline__ v16bf load_frag_b(const bf16_t* tileT, int ld,
                                                    int nbase, int kbase) {
  int lane = threadIdx.x & 31;
  const bf16_t* p = tileT + (size_t)(nbase + (lane & 15)) * ld + kbase + ((lane >> 4) << 3);
  v8bf lo = *(const v8bf*)p;
  v8bf hi = *(const v8bf*)(p + 16);
  return __builtin_shufflevector(lo, hi, 0,1,2,3,4,5,6,7,8,9,10,11,12,13,14,15);
}

static __device__ __forceinline__ v8f wmma_bf16(v16bf a, v16bf b, v8f c) {
  return __builtin_amdgcn_wmma_f32_16x16x32_bf16(false, a, false, b, (short)0, c,
                                                 false, false);
}

// ---------------------------------------------------------------------------
// Weight convert+transpose: W (KxN f32) -> Wt (NxK bf16)
// ---------------------------------------------------------------------------
__global__ __launch_bounds__(256) void wconv_kernel(const float* __restrict__ W,
                                                    bf16_t* __restrict__ Wt,
                                                    int K, int N) {
  __shared__ float tile[32][33];
  const int tid = threadIdx.x;
  const int nb = blockIdx.x * 32;
  const int kb = blockIdx.y * 32;
  #pragma unroll
  for (int i = tid; i < 1024; i += 256) {
    int k = i >> 5, n = i & 31;
    tile[k][n] = W[(size_t)(kb + k) * N + nb + n];
  }
  __syncthreads();
  #pragma unroll
  for (int i = tid; i < 1024; i += 256) {
    int n = i >> 5, k = i & 31;
    Wt[(size_t)(nb + n) * K + kb + k] = to_bf16(tile[k][n]);
  }
}

// ---------------------------------------------------------------------------
// x = q_embed + pos (f32 + bf16 copies); yb = bf16(qa_embed + pos)
// ---------------------------------------------------------------------------
__global__ __launch_bounds__(256) void addpos_kernel(
    const float* __restrict__ q, const float* __restrict__ qa,
    const float* __restrict__ pos, float* __restrict__ x,
    bf16_t* __restrict__ xb, bf16_t* __restrict__ yb) {
  size_t i = (size_t)blockIdx.x * 256 + threadIdx.x;
  size_t pi = i & (size_t)(SS * DDm - 1);
  float pv = pos[pi];
  float xv = q[i] + pv;
  float yv = qa[i] + pv;
  x[i]  = xv;
  xb[i] = to_bf16(xv);
  yb[i] = to_bf16(yv);
}

// ---------------------------------------------------------------------------
// bf16-WMMA GEMM: C = A(MxK bf16) @ Wt(NxK bf16)^T + bias, epilogue by mode.
// Tile 128x128, K-step 64, 256 threads (8 waves x [64 rows x 32 cols]).
// ---------------------------------------------------------------------------
#define MODE_RESID 1   // f32 out: acc + bias + resid(f32)
#define MODE_RELU  2   // bf16 out: relu(acc + bias)
#define MODE_KLAY  3   // bf16 out to (B,H,S,DK)
#define MODE_VLAY  4   // bf16 out to (B,H,DK,S)

__global__ __launch_bounds__(256) void gemm_kernel(
    const bf16_t* __restrict__ A, const bf16_t* __restrict__ Wt,
    const float* __restrict__ bias, const float* __restrict__ resid,
    void* __restrict__ Cout, int M, int N, int K, int mode) {
  __shared__ __align__(16) bf16_t As[128][64];
  __shared__ __align__(16) bf16_t Ws[128][64];   // N-major
  const int tid  = threadIdx.x;
  const int wave = tid >> 5;
  const int lane = tid & 31;
  const int m0 = blockIdx.x * 128;
  const int n0 = blockIdx.y * 128;
  const int wm = (wave >> 2) * 64;   // 0 / 64
  const int wn = (wave & 3) * 32;    // 0 / 32 / 64 / 96

  v8f acc[4][2] = {};
  const int nk = K >> 6;
  for (int kk = 0; kk < nk; ++kk) {
    const int kb = kk << 6;
    // 128x64 bf16 tiles = 1024 16B-units each; 4 units per thread per tile
    #pragma unroll
    for (int u = tid; u < 1024; u += 256) {
      int m = u >> 3, k8 = (u & 7) << 3;
      copy16_g2l(&As[m][k8], A + (size_t)(m0 + m) * K + kb + k8);
    }
    #pragma unroll
    for (int u = tid; u < 1024; u += 256) {
      int n = u >> 3, k8 = (u & 7) << 3;
      copy16_g2l(&Ws[n][k8], Wt + (size_t)(n0 + n) * K + kb + k8);
    }
    async_fence();
    __syncthreads();
    #pragma unroll
    for (int ks = 0; ks < 2; ++ks) {
      v16bf af[4], bf_[2];
      #pragma unroll
      for (int i = 0; i < 4; ++i) af[i] = load_frag_a(&As[0][0], 64, wm + 16 * i, ks * 32);
      #pragma unroll
      for (int j = 0; j < 2; ++j) bf_[j] = load_frag_b(&Ws[0][0], 64, wn + 16 * j, ks * 32);
      #pragma unroll
      for (int i = 0; i < 4; ++i)
        #pragma unroll
        for (int j = 0; j < 2; ++j)
          acc[i][j] = wmma_bf16(af[i], bf_[j], acc[i][j]);
    }
    __syncthreads();
  }

  const int cn = lane & 15;
  const int rh = (lane >> 4) * 8;
  float*  Cf = (float*)Cout;
  bf16_t* Cb = (bf16_t*)Cout;
  #pragma unroll
  for (int i = 0; i < 4; ++i) {
    #pragma unroll
    for (int j = 0; j < 2; ++j) {
      int gn = n0 + wn + 16 * j + cn;
      float bsv = bias[gn];
      #pragma unroll
      for (int r = 0; r < 8; ++r) {
        int gm = m0 + wm + 16 * i + rh + r;
        float v = acc[i][j][r] + bsv;
        if (mode == MODE_RESID) {
          Cf[(size_t)gm * N + gn] = v + resid[(size_t)gm * N + gn];
        } else if (mode == MODE_RELU) {
          Cb[(size_t)gm * N + gn] = to_bf16(fmaxf(v, 0.0f));
        } else if (mode == MODE_KLAY) {
          int b = gm >> 9, s = gm & 511, h = gn >> 6, dk = gn & 63;
          Cb[(((size_t)(b * HHn + h) * SS) + s) * DKK + dk] = to_bf16(v);
        } else { // MODE_VLAY
          int b = gm >> 9, s = gm & 511, h = gn >> 6, dk = gn & 63;
          Cb[(((size_t)(b * HHn + h) * DKK) + dk) * SS + s] = to_bf16(v);
        }
      }
    }
  }
}

// ---------------------------------------------------------------------------
// Attention: one block (4 waves) per (b, h, 16-row query block).
// scores = (qk.qkT)*scale*te*fr, strict causal mask, softmax, P@V.
// ---------------------------------------------------------------------------
__global__ __launch_bounds__(128) void attn_kernel(
    const bf16_t* __restrict__ qk,    // (B,H,S,DK) bf16
    const bf16_t* __restrict__ vT,    // (B,H,DK,S) bf16
    const float* __restrict__ pid,    // (B,S,S)
    const float* __restrict__ forget, // (B,S)
    bf16_t* __restrict__ ctx) {       // (B,S,D) bf16
  __shared__ __align__(16) bf16_t Qs[16][64];
  __shared__ __align__(16) bf16_t Ks[64][64];
  __shared__ __align__(16) bf16_t Pc[16][64];
  __shared__ __align__(16) bf16_t Vt[64][64];
  __shared__ float Sc[16][512];
  __shared__ float red[16][8];
  __shared__ float rowtmp[16];

  const int tid  = threadIdx.x;
  const int wave = tid >> 5;
  const int lane = tid & 31;
  const int blk  = blockIdx.x;
  const int it   = blk & 31;            // S/16 row blocks
  const int bh   = blk >> 5;
  const int b    = bh >> 3;
  const int h    = bh & 7;
  const int i0   = it << 4;
  const float scale = 0.125f;           // 1/sqrt(64)
  const bf16_t* qbase = qk + (size_t)bh * SS * DKK;

  // Q tile: 16x64 bf16 = 128 16B-units, one per thread
  {
    int u = tid;
    int m = u >> 3, k8 = (u & 7) << 3;
    copy16_g2l(&Qs[m][k8], qbase + (size_t)(i0 + m) * DKK + k8);
  }

  const int nch = (i0 >> 6) + 1;        // 64-key chunks covering j <= i0+14

  // ---- scores ----
  for (int c = 0; c < nch; ++c) {
    const int j0 = c << 6;
    #pragma unroll
    for (int u = tid; u < 512; u += 128) {
      int j = u >> 3, k8 = (u & 7) << 3;
      copy16_g2l(&Ks[j][k8], qbase + (size_t)(j0 + j) * DKK + k8);
    }
    async_fence();
    __syncthreads();
    const int jb = j0 + (wave << 4);
    if (jb <= i0 + 14) {
      v8f acc = {};
      acc = wmma_bf16(load_frag_a(&Qs[0][0], 64, 0, 0),
                      load_frag_b(&Ks[0][0], 64, wave << 4, 0), acc);
      acc = wmma_bf16(load_frag_a(&Qs[0][0], 64, 0, 32),
                      load_frag_b(&Ks[0][0], 64, wave << 4, 32), acc);
      const int j   = jb + (lane & 15);
      const int rh  = (lane >> 4) * 8;
      const float frj = forget[b * SS + j];
      #pragma unroll
      for (int r = 0; r < 8; ++r) {
        const int gi = i0 + rh + r;
        float sv;
        if (j < gi) {
          float p  = pid[((size_t)b * SS + gi) * SS + j];
          float sg = 1.0f / (1.0f + __expf(-p));
          float te = __expf(sg);
          te = fminf(fmaxf(te, 1e-5f), 1e5f);
          sv = acc[r] * scale * te * frj;
        } else {
          sv = NEGV;
        }
        Sc[rh + r][j] = sv;
      }
    }
    __syncthreads();
  }

  // ---- softmax over valid prefix (j < gi) ----
  {
    const int r = tid >> 3, sub = tid & 7;
    const int gi = i0 + r;
    float mx = -3.0e38f;
    for (int j = sub; j < gi; j += 8) mx = fmaxf(mx, Sc[r][j]);
    red[r][sub] = mx;
    __syncthreads();
    if (sub == 0) {
      float m2 = red[r][0];
      #pragma unroll
      for (int q = 1; q < 8; ++q) m2 = fmaxf(m2, red[r][q]);
      rowtmp[r] = m2;
    }
    __syncthreads();
    const float rm = rowtmp[r];
    float sm = 0.0f;
    for (int j = sub; j < gi; j += 8) {
      float e = __expf(Sc[r][j] - rm);
      Sc[r][j] = e;
      sm += e;
    }
    red[r][sub] = sm;
    __syncthreads();
    if (sub == 0) {
      float s2 = 0.0f;
      #pragma unroll
      for (int q = 0; q < 8; ++q) s2 += red[r][q];
      rowtmp[r] = (gi > 0 && s2 > 0.0f) ? (1.0f / s2) : 0.0f;  // row 0 -> zeroed
    }
    __syncthreads();
  }

  // ---- P @ V ----  (each wave owns one 16-wide e-tile)
  v8f oacc = {};
  const int eb = wave << 4;
  const bf16_t* vbase = vT + (size_t)bh * DKK * SS;
  for (int c = 0; c < nch; ++c) {
    const int j0 = c << 6;
    #pragma unroll
    for (int u = tid; u < 512; u += 128) {
      int e = u >> 3, j8 = (u & 7) << 3;
      copy16_g2l(&Vt[e][j8], vbase + (size_t)e * SS + j0 + j8);
    }
    #pragma unroll
    for (int i = tid; i < 16 * 64; i += 128) {
      int m = i >> 6, jl = i & 63;
      int j = j0 + jl;
      float pv = (j < i0 + m) ? Sc[m][j] * rowtmp[m] : 0.0f;
      Pc[m][jl] = to_bf16(pv);
    }
    async_fence();
    __syncthreads();
    oacc = wmma_bf16(load_frag_a(&Pc[0][0], 64, 0, 0),
                     load_frag_b(&Vt[0][0], 64, eb, 0), oacc);
    oacc = wmma_bf16(load_frag_a(&Pc[0][0], 64, 0, 32),
                     load_frag_b(&Vt[0][0], 64, eb, 32), oacc);
    __syncthreads();
  }

  const int cn  = lane & 15;
  const int rh2 = (lane >> 4) * 8;
  bf16_t* cbase = ctx + ((size_t)b * SS + i0) * DDm + h * DKK;
  #pragma unroll
  for (int r = 0; r < 8; ++r)
    cbase[(size_t)(rh2 + r) * DDm + eb + cn] = to_bf16(oacc[r]);
}

// ---------------------------------------------------------------------------
// Row LayerNorm over D=512: x (f32, in place) and bf16 copy xb.
// ---------------------------------------------------------------------------
__global__ __launch_bounds__(128) void ln_kernel(float* __restrict__ x,
                                                 bf16_t* __restrict__ xb,
                                                 const float* __restrict__ g,
                                                 const float* __restrict__ bb) {
  __shared__ float red[128];
  const int tid = threadIdx.x;
  float*  p  = x  + (size_t)blockIdx.x * DDm;
  bf16_t* pb = xb + (size_t)blockIdx.x * DDm;
  float v[4];
  float s = 0.0f;
  #pragma unroll
  for (int i = 0; i < 4; ++i) { v[i] = p[tid + 128 * i]; s += v[i]; }
  red[tid] = s;
  __syncthreads();
  for (int off = 64; off > 0; off >>= 1) {
    if (tid < off) red[tid] += red[tid + off];
    __syncthreads();
  }
  const float mean = red[0] * (1.0f / DDm);
  __syncthreads();
  s = 0.0f;
  #pragma unroll
  for (int i = 0; i < 4; ++i) { float d = v[i] - mean; s += d * d; }
  red[tid] = s;
  __syncthreads();
  for (int off = 64; off > 0; off >>= 1) {
    if (tid < off) red[tid] += red[tid + off];
    __syncthreads();
  }
  const float inv = rsqrtf(red[0] * (1.0f / DDm) + 1e-5f);
  #pragma unroll
  for (int i = 0; i < 4; ++i) {
    int c = tid + 128 * i;
    float o = (v[i] - mean) * inv * g[c] + bb[c];
    p[c]  = o;
    pb[c] = to_bf16(o);
  }
}

// ---------------------------------------------------------------------------
extern "C" void kernel_launch(void* const* d_in, const int* in_sizes, int n_in,
                              void* d_out, int out_size, void* d_ws, size_t ws_size,
                              hipStream_t stream) {
  const float* q_emb  = (const float*)d_in[0];
  const float* qa_emb = (const float*)d_in[1];
  const float* forget = (const float*)d_in[2];   // (B,S,1)
  const float* pid    = (const float*)d_in[3];   // (B,S,S)
  const float* pos    = (const float*)d_in[4];   // (1,S,D)
  const float* Wk  = (const float*)d_in[5];
  const float* bk  = (const float*)d_in[6];
  const float* Wv  = (const float*)d_in[7];
  const float* bv  = (const float*)d_in[8];
  const float* Wo  = (const float*)d_in[9];
  const float* bo  = (const float*)d_in[10];
  const float* l1g = (const float*)d_in[11];
  const float* l1b = (const float*)d_in[12];
  const float* W1  = (const float*)d_in[13];
  const float* b1  = (const float*)d_in[14];
  const float* W2  = (const float*)d_in[15];
  const float* b2  = (const float*)d_in[16];
  const float* l2g = (const float*)d_in[17];
  const float* l2b = (const float*)d_in[18];

  const size_t NBSD = (size_t)BB * SS * DDm;     // 8,388,608
  char* wsb = (char*)d_ws;
  size_t off = 0;
  float*  X    = (float*)(wsb + off);  off += NBSD * 4;
  bf16_t* Xb   = (bf16_t*)(wsb + off); off += NBSD * 2;
  bf16_t* Yb   = (bf16_t*)(wsb + off); off += NBSD * 2;
  bf16_t* QKb  = (bf16_t*)(wsb + off); off += NBSD * 2;
  bf16_t* VTb  = (bf16_t*)(wsb + off); off += NBSD * 2;
  bf16_t* CTXb = (bf16_t*)(wsb + off); off += NBSD * 2;
  bf16_t* HIDb = (bf16_t*)(wsb + off); off += (size_t)MROWS * FFF * 2;
  bf16_t* WkT  = (bf16_t*)(wsb + off); off += (size_t)LLn * DDm * DDm * 2;
  bf16_t* WvT  = (bf16_t*)(wsb + off); off += (size_t)LLn * DDm * DDm * 2;
  bf16_t* WoT  = (bf16_t*)(wsb + off); off += (size_t)LLn * DDm * DDm * 2;
  bf16_t* W1T  = (bf16_t*)(wsb + off); off += (size_t)LLn * DDm * FFF * 2;
  bf16_t* W2T  = (bf16_t*)(wsb + off); off += (size_t)LLn * FFF * DDm * 2;

  // --- one-shot weight convert+transpose (K x N f32 -> N x K bf16) ---
  dim3 tDD(DDm / 32, DDm / 32);     // 16x16
  dim3 t1(FFF / 32, DDm / 32);      // W1: K=D, N=FF -> grid (N/32, K/32)
  dim3 t2(DDm / 32, FFF / 32);      // W2: K=FF, N=D
  for (int l = 0; l < LLn; ++l) {
    const size_t wdd = (size_t)l * DDm * DDm;
    const size_t wdf = (size_t)l * DDm * FFF;
    wconv_kernel<<<tDD, 256, 0, stream>>>(Wk + wdd, WkT + wdd, DDm, DDm);
    wconv_kernel<<<tDD, 256, 0, stream>>>(Wv + wdd, WvT + wdd, DDm, DDm);
    wconv_kernel<<<tDD, 256, 0, stream>>>(Wo + wdd, WoT + wdd, DDm, DDm);
    wconv_kernel<<<t1, 256, 0, stream>>>(W1 + wdf, W1T + wdf, DDm, FFF);
    wconv_kernel<<<t2, 256, 0, stream>>>(W2 + wdf, W2T + wdf, FFF, DDm);
  }

  addpos_kernel<<<(unsigned)(NBSD / 256), 256, 0, stream>>>(q_emb, qa_emb, pos,
                                                            X, Xb, Yb);

  dim3 gD(MROWS / 128, DDm / 128);    // 128 x 4
  dim3 gF1(MROWS / 128, FFF / 128);   // 128 x 16
  const unsigned gAttn = (unsigned)(BB * HHn * (SS / 16));  // 8192

  for (int l = 0; l < LLn; ++l) {
    const size_t wdd = (size_t)l * DDm * DDm;
    const size_t wdf = (size_t)l * DDm * FFF;
    gemm_kernel<<<gD, 256, 0, stream>>>(Xb, WkT + wdd, bk + l * DDm, nullptr, QKb,
                                        MROWS, DDm, DDm, MODE_KLAY);
    gemm_kernel<<<gD, 256, 0, stream>>>(Yb, WvT + wdd, bv + l * DDm, nullptr, VTb,
                                        MROWS, DDm, DDm, MODE_VLAY);
    attn_kernel<<<gAttn, 128, 0, stream>>>(QKb, VTb, pid, forget, CTXb);
    gemm_kernel<<<gD, 256, 0, stream>>>(CTXb, WoT + wdd, bo + l * DDm, X, X,
                                        MROWS, DDm, DDm, MODE_RESID);
    ln_kernel<<<MROWS, 128, 0, stream>>>(X, Xb, l1g + l * DDm, l1b + l * DDm);
    gemm_kernel<<<gF1, 256, 0, stream>>>(Xb, W1T + wdf, b1 + l * FFF, nullptr, HIDb,
                                         MROWS, FFF, DDm, MODE_RELU);
    gemm_kernel<<<gD, 256, 0, stream>>>(HIDb, W2T + wdf, b2 + l * DDm, X, X,
                                        MROWS, DDm, FFF, MODE_RESID);
    ln_kernel<<<MROWS, 128, 0, stream>>>(X, Xb, l2g + l * DDm, l2b + l * DDm);
  }

  (void)hipMemcpyAsync(d_out, X, NBSD * sizeof(float), hipMemcpyDeviceToDevice, stream);
}
